// GCN_dropedge_53008486367825
// MI455X (gfx1250) — matrix operand order repaired and checked by reading, not compiled, not run on hardware
//
#include <hip/hip_runtime.h>
#include <hip/hip_bf16.h>

// ---------------------------------------------------------------------------
// Types for CDNA5 WMMA (gfx1250, wave32)
// ---------------------------------------------------------------------------
typedef __bf16 bf16_t;
typedef __attribute__((ext_vector_type(16))) __bf16 v16bf;
typedef __attribute__((ext_vector_type(8)))  float  v8f;

__device__ __forceinline__ float4 relu4(float4 v) {
    v.x = fmaxf(v.x, 0.f); v.y = fmaxf(v.y, 0.f);
    v.z = fmaxf(v.z, 0.f); v.w = fmaxf(v.w, 0.f);
    return v;
}

// ---------------------------------------------------------------------------
// Degree / renormalization kernels
// ---------------------------------------------------------------------------
__global__ void deg_init_kernel(float* __restrict__ deg, int N) {
    int i = blockIdx.x * blockDim.x + threadIdx.x;
    if (i < N) deg[i] = 1.0f;   // identity self-loop contributes 1
}

__global__ void deg_accum_kernel(const int* __restrict__ row,
                                 const float* __restrict__ vals,
                                 const int* __restrict__ mask,
                                 float* __restrict__ deg, int E) {
    int e = blockIdx.x * blockDim.x + threadIdx.x;
    if (e >= E) return;
    if (!mask[e]) return;                     // dropped edge -> value 0
    unsafeAtomicAdd(&deg[row[e]], vals[e]);   // native global_atomic_add_f32
}

__global__ void dis_kernel(const float* __restrict__ deg,
                           float* __restrict__ dis, int N) {
    int i = blockIdx.x * blockDim.x + threadIdx.x;
    if (i < N) dis[i] = fminf(rsqrtf(deg[i]), 10.0f);  // clip(deg^-1/2, 0, 10)
}

// ---------------------------------------------------------------------------
// Weight pre-swizzle: f32 W[128 x DN] -> bf16 fragments laid out so that each
// wave's B fragment (per (tn, kb) pair) is 32 contiguous bytes per lane.
//   frag = tn*4 + kb/32 ; element index = (frag*32 + lane)*16 + j
//   value = W[kb + 16*(lane>>4) + j][tn*16 + (lane&15)]
// ---------------------------------------------------------------------------
template<int DN>
__global__ void wswizzle_kernel(const float* __restrict__ W,
                                bf16_t* __restrict__ Wb) {
    const int idx = blockIdx.x * blockDim.x + threadIdx.x;
    if (idx >= DN * 128) return;
    const int j    = idx & 15;
    const int lane = (idx >> 4) & 31;
    const int frag = idx >> 9;          // 512 elements per fragment
    const int kbi  = frag & 3;
    const int tn   = frag >> 2;
    const int half = lane >> 4;
    const int l15  = lane & 15;
    const int k = kbi * 32 + half * 16 + j;
    const int n = tn * 16 + l15;
    Wb[idx] = (bf16_t)W[(size_t)k * DN + n];
}

// ---------------------------------------------------------------------------
// bf16 WMMA GEMM:  C[M x DN] = A[M x 128] * W[128 x DN]
// One wave computes TWO 16x16 tiles (rows m0..m0+15 and m0+16..m0+31) for
// column tile `wave`, reusing each B fragment twice.
// blockDim.x = (DN/16)*32, gridDim.x = ceil(M/32).
// A fragment (16x32 bf16): a[j] = A[m][kb+16h+j]
// C/D (16x16 f32): c[i] = C[8h+i][n]
// ---------------------------------------------------------------------------
template<bool RELU_A>
__device__ __forceinline__ v16bf load_a_frag(const float4* a4, int kb, int half) {
    float4 p0 = a4[(kb + half * 16) / 4 + 0];
    float4 p1 = a4[(kb + half * 16) / 4 + 1];
    float4 p2 = a4[(kb + half * 16) / 4 + 2];
    float4 p3 = a4[(kb + half * 16) / 4 + 3];
    if (RELU_A) { p0 = relu4(p0); p1 = relu4(p1); p2 = relu4(p2); p3 = relu4(p3); }
    v16bf a;
    a[0]  = (bf16_t)p0.x; a[1]  = (bf16_t)p0.y; a[2]  = (bf16_t)p0.z; a[3]  = (bf16_t)p0.w;
    a[4]  = (bf16_t)p1.x; a[5]  = (bf16_t)p1.y; a[6]  = (bf16_t)p1.z; a[7]  = (bf16_t)p1.w;
    a[8]  = (bf16_t)p2.x; a[9]  = (bf16_t)p2.y; a[10] = (bf16_t)p2.z; a[11] = (bf16_t)p2.w;
    a[12] = (bf16_t)p3.x; a[13] = (bf16_t)p3.y; a[14] = (bf16_t)p3.z; a[15] = (bf16_t)p3.w;
    return a;
}

template<int DN, bool RELU_A>
__global__ void gemm_bf16_wmma_kernel(const float* __restrict__ A,
                                      const bf16_t* __restrict__ Wb,
                                      float* __restrict__ C, int M) {
    constexpr int K = 128;
    const int wave = threadIdx.x >> 5;        // column tile index
    const int lane = threadIdx.x & 31;
    const int l15  = lane & 15;
    const int half = lane >> 4;
    const int m0   = blockIdx.x * 32;
    const int m1   = m0 + 16;
    const int n    = wave * 16 + l15;

    // Clamp second-tile row for loads (keeps EXEC all-ones for WMMA);
    // stores for the second tile are guarded by a wave-uniform branch.
    const int r0 = m0 + l15;
    const int r1 = (m1 + l15 < M) ? (m1 + l15) : (M - 1);
    const float4* a4_0 = (const float4*)(A + (size_t)r0 * K);
    const float4* a4_1 = (const float4*)(A + (size_t)r1 * K);

    v8f acc0 = {}, acc1 = {};
#pragma unroll
    for (int kb = 0; kb < K; kb += 32) {
        // B fragment: one contiguous 32-byte load per lane
        const v16bf b = *(const v16bf*)(Wb + ((size_t)(wave * 4 + (kb >> 5)) * 32 + lane) * 16);
        const v16bf a0 = load_a_frag<RELU_A>(a4_0, kb, half);
        const v16bf a1 = load_a_frag<RELU_A>(a4_1, kb, half);
        acc0 = __builtin_amdgcn_wmma_f32_16x16x32_bf16(
                   false, a0, false, b, (short)0, acc0, false, false);
        acc1 = __builtin_amdgcn_wmma_f32_16x16x32_bf16(
                   false, a1, false, b, (short)0, acc1, false, false);
    }

#pragma unroll
    for (int i = 0; i < 8; ++i)
        C[(size_t)(m0 + half * 8 + i) * DN + n] = acc0[i];
    if (m1 < M) {
#pragma unroll
        for (int i = 0; i < 8; ++i)
            C[(size_t)(m1 + half * 8 + i) * DN + n] = acc1[i];
    }
}

// ---------------------------------------------------------------------------
// SpMM:  out = D^-1/2 (A+I) D^-1/2  applied to `in`
// Self-loop term initializes out (also covers poisoned buffers).
// ---------------------------------------------------------------------------
__global__ void spmm_self_kernel(const float* __restrict__ in,
                                 const float* __restrict__ dis,
                                 float* __restrict__ out, int D) {
    const int i = blockIdx.x;
    const int f = threadIdx.x;
    const float s = dis[i];
    out[(size_t)i * D + f] = s * s * in[(size_t)i * D + f];
}

// One wave per edge; lane handles D/32 contiguous features (vector gather).
template<int D>
__global__ void spmm_edge_kernel(const int* __restrict__ row,
                                 const int* __restrict__ col,
                                 const float* __restrict__ vals,
                                 const int* __restrict__ mask,
                                 const float* __restrict__ dis,
                                 const float* __restrict__ in,
                                 float* __restrict__ out, int E) {
    const int e = blockIdx.x * (blockDim.x >> 5) + (threadIdx.x >> 5);
    if (e >= E) return;
    if (!mask[e]) return;                        // wave-uniform early out
    const int r = row[e];
    const int c = col[e];
    const float nv = vals[e] * dis[r] * dis[c];
    const int lane = threadIdx.x & 31;
    if constexpr (D == 128) {
        const float4 v = ((const float4*)(in + (size_t)c * D))[lane];   // b128 gather
        float* dst = out + (size_t)r * D + lane * 4;
        unsafeAtomicAdd(dst + 0, nv * v.x);
        unsafeAtomicAdd(dst + 1, nv * v.y);
        unsafeAtomicAdd(dst + 2, nv * v.z);
        unsafeAtomicAdd(dst + 3, nv * v.w);
    } else {
        const float2 v = ((const float2*)(in + (size_t)c * D))[lane];   // b64 gather
        float* dst = out + (size_t)r * D + lane * 2;
        unsafeAtomicAdd(dst + 0, nv * v.x);
        unsafeAtomicAdd(dst + 1, nv * v.y);
    }
}

// ---------------------------------------------------------------------------
// Host-side launch
// ---------------------------------------------------------------------------
extern "C" void kernel_launch(void* const* d_in, const int* in_sizes, int n_in,
                              void* d_out, int out_size, void* d_ws, size_t ws_size,
                              hipStream_t stream) {
    (void)n_in; (void)out_size; (void)ws_size;

    const float* x     = (const float*)d_in[0];   // [N,128]
    const int*   erow  = (const int*)  d_in[1];   // [E]
    const int*   ecol  = (const int*)  d_in[2];   // [E]
    const float* evals = (const float*)d_in[3];   // [E]
    const int*   emask = (const int*)  d_in[4];   // [E] bernoulli keep mask
    const float* W0    = (const float*)d_in[5];   // [128,128]
    const float* W1    = (const float*)d_in[6];   // [128,64]
    float*       out   = (float*)d_out;           // [N,64]

    const int D_IN = 128, D_HID = 128, D_OUT = 64;
    const int N = in_sizes[0] / D_IN;             // 50000
    const int E = in_sizes[1];                    // 600000

    // Workspace carving (256B aligned slices)
    char*  ws  = (char*)d_ws;
    size_t off = 0;
    auto carve = [&](size_t bytes) -> void* {
        void* p = ws + off;
        off += (bytes + 255) & ~(size_t)255;
        return p;
    };
    float*  deg  = (float*)carve((size_t)N * sizeof(float));
    float*  dis  = (float*)carve((size_t)N * sizeof(float));
    bf16_t* Wb0  = (bf16_t*)carve((size_t)D_IN  * D_HID * sizeof(bf16_t)); // 32KB
    bf16_t* Wb1  = (bf16_t*)carve((size_t)D_HID * D_OUT * sizeof(bf16_t)); // 16KB
    float*  bufA = (float*)carve((size_t)N * D_HID * sizeof(float)); // XW0, reused for HW1
    float*  bufH = (float*)carve((size_t)N * D_HID * sizeof(float)); // spmm1 output

    // 0) weight bf16 swizzle (fragment layout)
    wswizzle_kernel<128><<<(128 * 128 + 255) / 256, 256, 0, stream>>>(W0, Wb0);
    wswizzle_kernel<64> <<<(128 * 64  + 255) / 256, 256, 0, stream>>>(W1, Wb1);

    // 1) degree + D^{-1/2}
    deg_init_kernel <<<(N + 255) / 256, 256, 0, stream>>>(deg, N);
    deg_accum_kernel<<<(E + 255) / 256, 256, 0, stream>>>(erow, evals, emask, deg, E);
    dis_kernel      <<<(N + 255) / 256, 256, 0, stream>>>(deg, dis, N);

    // 2) XW0 = x @ W0  (bf16 WMMA, f32 accumulate; 2 M-tiles per wave)
    gemm_bf16_wmma_kernel<128, false><<<(N + 31) / 32, 8 * 32, 0, stream>>>(x, Wb0, bufA, N);

    // 3) H_pre = support @ XW0   (relu deferred into GEMM2's A-load)
    spmm_self_kernel<<<N, D_HID, 0, stream>>>(bufA, dis, bufH, D_HID);
    spmm_edge_kernel<128><<<(E + 7) / 8, 256, 0, stream>>>(erow, ecol, evals, emask,
                                                           dis, bufA, bufH, E);

    // 4) HW1 = relu(H_pre) @ W1  (reuse bufA)
    gemm_bf16_wmma_kernel<64, true><<<(N + 31) / 32, 4 * 32, 0, stream>>>(bufH, Wb1, bufA, N);

    // 5) out = support @ HW1
    spmm_self_kernel<<<N, D_OUT, 0, stream>>>(bufA, dis, out, D_OUT);
    spmm_edge_kernel<64><<<(E + 7) / 8, 256, 0, stream>>>(erow, ecol, evals, emask,
                                                          dis, bufA, out, E);
}
